// InnerShiftTriple_91156385890481
// MI455X (gfx1250) — matrix operand order repaired and checked by reading, compiled, not run on hardware
//
#include <hip/hip_runtime.h>

#define B_ 8
#define C_ 512
#define C2 256
#define HW 4096
#define W_ 64
#define PFLAG 1024
#define QTILES_TOTAL 256
#define QSPLIT 4
#define QT_PER_WG (QTILES_TOTAL / QSPLIT)   // 64 q-tiles of 16 per WG
#define WAVES 4
#define THREADS (WAVES * 32)

typedef __attribute__((ext_vector_type(16))) __bf16 v16bf;
typedef __attribute__((ext_vector_type(8)))  float  v8f;
typedef __attribute__((ext_vector_type(4)))  unsigned int u32x4;
typedef __attribute__((ext_vector_type(8)))  int    i32x8;
typedef __attribute__((ext_vector_type(4)))  int    i32x4;

__device__ __forceinline__ __bf16 f2bf(float f) {
    union { float f; unsigned u; } v; v.f = f;
    unsigned r = v.u + 0x7FFFu + ((v.u >> 16) & 1u);   // round-to-nearest-even
    unsigned short h = (unsigned short)(r >> 16);
    union { unsigned short s; __bf16 b; } o; o.s = h;
    return o.b;
}

// ---------------- rnorm[b][p] = 1 / max(||latter[b,:,p]||, 1e-8) ----------------
__global__ void norm_kernel(const float* __restrict__ x, float* __restrict__ rnorm) {
    int t = blockIdx.x * blockDim.x + threadIdx.x;      // B_*HW threads
    int b = t / HW, p = t - b * HW;
    const float* lf = x + (size_t)b * C_ * HW + (size_t)C2 * HW + p;
    float s = 0.f;
    #pragma unroll 8
    for (int k = 0; k < C2; ++k) { float v = lf[(size_t)k * HW]; s += v * v; }
    rnorm[t] = 1.0f / fmaxf(sqrtf(s), 1e-8f);
}

// ------- lnbf[b][q][k] = bf16(latter[b][k][q] * rnorm[b][q])  (q-major transpose) -------
// block: 256 threads handles one b and 32 consecutive q (32q x 256k tile via LDS)
__global__ void prep_kernel(const float* __restrict__ x, const float* __restrict__ rnorm,
                            __bf16* __restrict__ lnbf) {
    int b  = blockIdx.x >> 7;
    int q0 = (blockIdx.x & 127) * 32;
    __shared__ __bf16 T[32][260];
    const float* lf = x + ((size_t)b * C_ + C2) * HW;
    const float* rn = rnorm + (size_t)b * HW;
    #pragma unroll
    for (int i = 0; i < 32; ++i) {                      // read coalesced over q
        int f = threadIdx.x + i * 256;
        int k = f >> 5, qq = f & 31;
        int q = q0 + qq;
        T[qq][k] = f2bf(lf[(size_t)k * HW + q] * rn[q]);
    }
    __syncthreads();
    __bf16* dst = lnbf + ((size_t)b * HW + q0) * C2;
    #pragma unroll
    for (int i = 0; i < 32; ++i) {                      // write coalesced over k
        int f = threadIdx.x + i * 256;
        int qq = f >> 8, k = f & 255;
        dst[(size_t)qq * C2 + k] = T[qq][k];
    }
}

// ---------------- out[:, 0:512, :] = input (float4 streaming copy) ----------------
__global__ void copy_kernel(const float* __restrict__ x, float* __restrict__ out) {
    int i = blockIdx.x * blockDim.x + threadIdx.x;      // B_*C_*HW/4 float4s
    const int per_b = C_ * HW / 4;
    int b = i / per_b, r = i - b * per_b;
    const float4* src = (const float4*)x;
    float4* dst = (float4*)out;
    dst[(size_t)b * ((C_ + C2) * HW / 4) + r] = src[(size_t)b * per_b + r];
}

// ---------------- WMMA cos-sim + streaming argmax over a q slice ----------------
// grid: B_ * 16 * QSPLIT blocks of 128 threads (4 waves); wave -> one 16-row p-tile.
// B panels (16q x 256k bf16, contiguous 8KB in lnbf) move via the Tensor Data Mover
// into double-buffered LDS; TDM pad feature reproduces the 264-element padded rows.
__launch_bounds__(THREADS)
__global__ void cos_argmax_kernel(const float* __restrict__ x,
                                  const int*   __restrict__ flag,
                                  const __bf16* __restrict__ lnbf,
                                  float* __restrict__ pmax,
                                  int*   __restrict__ pidx) {
    const int qs  = blockIdx.x & (QSPLIT - 1);
    const int ptg = (blockIdx.x / QSPLIT) & 15;
    const int b   = blockIdx.x / (QSPLIT * 16);
    const int wave = threadIdx.x >> 5;
    const int lane = threadIdx.x & 31;
    const int ln = lane & 15;          // N column (B/C/D) or M row (A)
    const int lh = lane >> 4;          // lane-half select

    __shared__ __bf16 Bs[2][16][264];  // row stride 528 B (16B aligned for b128 reads)

    const int q0base = qs * QT_PER_WG * 16;
    const __bf16* lb = lnbf + (size_t)b * HW * C2;

    // ---- TDM panel issue: 2D tile 256(k) x 16(q), 2B elems, LDS row pad 512B->528B ----
    auto issue_panel = [&](int jt, int buf) {
        unsigned long long gaddr =
            (unsigned long long)(const void*)(lb + (size_t)(q0base + jt * 16) * C2);
        unsigned ldsa = (unsigned)(unsigned long long)(const void*)&Bs[buf][0][0];
        u32x4 g0;
        g0[0] = 1u;                                    // count=1 user descriptor
        g0[1] = ldsa;                                  // lds_addr (bytes)
        g0[2] = (unsigned)gaddr;                       // global_addr[31:0]
        g0[3] = (unsigned)((gaddr >> 32) & 0x01FFFFFFull) | (2u << 30);  // [56:32] | type=2
        i32x8 g1;
        g1[0] = (1 << 16)                              // data_size = 2 bytes
              | (1 << 20)                              // pad_enable
              | (6 << 22)                              // pad_interval: 128 DWORDs (one row)
              | (3 << 25);                             // pad_amount: 4 DWORDs (16 B)
        g1[1] = (int)(256u << 16);                     // tensor_dim0 = 256 (low16)
        g1[2] = (int)(4096u << 16);                    // tensor_dim1 = 4096 (low16)
        g1[3] = (int)(256u << 16);                     // tile_dim0 = 256 (k)
        g1[4] = 16;                                    // tile_dim1 = 16 (q), tile_dim2 = 0
        g1[5] = 256;                                   // tensor_dim0_stride = 256
        g1[6] = 0; g1[7] = 0;
        i32x4 gz4 = {0, 0, 0, 0};
        i32x8 gz8 = {0, 0, 0, 0, 0, 0, 0, 0};
        __builtin_amdgcn_tensor_load_to_lds(g0, g1, gz4, gz4, gz8, 0);
    };

    if (wave == 0) issue_panel(0, 0);                  // kick panel 0 (overlaps A loads)

    // ---- A fragments: 16 flagged p-rows x 256 K of `former`, resident for whole sweep ----
    // ISA 7.12.2 16-bit A 16x32: lanes0-15 K=[0..7]|[16..23], lanes16-31 K=[8..15]|[24..31]
    const float* ff = x + (size_t)b * C_ * HW;
    const int pt = ptg * WAVES + wave;                 // p-tile 0..63
    const int pi = pt * 16 + ln;                       // flagged index 0..1023
    const int p  = (16 + (pi >> 5)) * W_ + 16 + (pi & 31);
    v16bf a[8];
    #pragma unroll
    for (int kk = 0; kk < 8; ++kk) {
        #pragma unroll
        for (int e = 0; e < 16; ++e) {
            int K = kk * 32 + ((e & 8) << 1) + (lh << 3) + (e & 7);
            a[kk][e] = f2bf(ff[(size_t)K * HW + p]);
        }
    }

    if (wave == 0) __builtin_amdgcn_s_wait_tensorcnt(0);
    __syncthreads();

    float rmax[8]; int ridx[8];
    #pragma unroll
    for (int v = 0; v < 8; ++v) { rmax[v] = -__builtin_inff(); ridx[v] = 0; }

    for (int jt = 0; jt < QT_PER_WG; ++jt) {
        const int cur = jt & 1;
        const bool do_next = (jt + 1) < QT_PER_WG;
        if (do_next && wave == 0) issue_panel(jt + 1, cur ^ 1);   // async prefetch

        // Load all 8 B fragments into distinct registers first (16 back-to-back
        // ds_load_b128), so the WMMA chain overlaps LDS latency instead of
        // stalling on s_wait_dscnt 0 per fragment.
        v16bf bfr[8];
        #pragma unroll
        for (int kk = 0; kk < 8; ++kk) {
            int kb = kk * 32 + (lh << 3);              // B 32x16: lanes0-15 K=[0..7]|[16..23], lanes16-31 K=[8..15]|[24..31]
            #pragma unroll
            for (int e = 0; e < 16; ++e) bfr[kk][e] = Bs[cur][ln][kb + ((e & 8) << 1) + (e & 7)];
        }

        // 16x16 cos tile: 8 x v_wmma_f32_16x16x32_bf16 over K=256
        v8f acc = {};
        #pragma unroll
        for (int kk = 0; kk < 8; ++kk) {
            acc = __builtin_amdgcn_wmma_f32_16x16x32_bf16(
                false, a[kk], false, bfr[kk], (short)0, acc, false, false);
        }

        // streaming argmax with flag mask (flag==1 -> -inf per reference)
        {
            int q = q0base + jt * 16 + ln;
            bool ok = (flag[q] == 0);
            #pragma unroll
            for (int v = 0; v < 8; ++v) {
                float c = acc[v];
                if (ok && c > rmax[v]) { rmax[v] = c; ridx[v] = q; }
            }
        }

        if (do_next && wave == 0) __builtin_amdgcn_s_wait_tensorcnt(0);
        __syncthreads();
    }

    // reduce across the 16 lanes of each half (C/D layout: VGPR v -> row M=v+8*lh)
    #pragma unroll
    for (int v = 0; v < 8; ++v) {
        float m = rmax[v]; int ix = ridx[v];
        #pragma unroll
        for (int off = 8; off > 0; off >>= 1) {
            float om = __shfl_xor(m, off, 32);
            int   oi = __shfl_xor(ix, off, 32);
            if (om > m || (om == m && oi < ix)) { m = om; ix = oi; }
        }
        if (ln == 0) {
            int piOut = pt * 16 + v + (lh << 3);
            int slot = (b * PFLAG + piOut) * QSPLIT + qs;
            pmax[slot] = m; pidx[slot] = ix;
        }
    }
}

// ---------------- merge the QSPLIT partial argmaxes ----------------
__global__ void argmax_merge_kernel(const float* __restrict__ pmax,
                                    const int*   __restrict__ pidx,
                                    int* __restrict__ idxout) {
    int t = blockIdx.x * blockDim.x + threadIdx.x;
    if (t >= B_ * PFLAG) return;
    float best = -__builtin_inff(); int bi = 0;
    #pragma unroll
    for (int s = 0; s < QSPLIT; ++s) {
        float m = pmax[t * QSPLIT + s]; int ix = pidx[t * QSPLIT + s];
        if (m > best || (m == best && ix < bi)) { best = m; bi = ix; }
    }
    idxout[t] = bi;
}

// ---------------- shift gather: out[:, 512:768, :] ----------------
__global__ void gather_kernel(const float* __restrict__ x,
                              const int*   __restrict__ flag,
                              const int*   __restrict__ idx,
                              float* __restrict__ out) {
    int t = blockIdx.x * blockDim.x + threadIdx.x;      // B_*C2*HW
    int p = t & (HW - 1);
    int c = (t >> 12) & (C2 - 1);
    int b = t >> 20;
    float v = 0.f;
    if (flag[p] == 1) {
        int h = p >> 6, w = p & 63;
        int piF = ((h - 16) << 5) + (w - 16);
        int j = idx[b * PFLAG + piF];
        v = x[(size_t)b * C_ * HW + (size_t)(C2 + c) * HW + j];
    }
    out[(size_t)b * (C_ + C2) * HW + (size_t)(C_ + c) * HW + p] = v;
}

extern "C" void kernel_launch(void* const* d_in, const int* in_sizes, int n_in,
                              void* d_out, int out_size, void* d_ws, size_t ws_size,
                              hipStream_t stream) {
    const float* x    = (const float*)d_in[0];
    const int*   flag = (const int*)d_in[1];
    float* out = (float*)d_out;

    char* ws = (char*)d_ws;
    __bf16* lnbf = (__bf16*)ws;                          // 8*4096*256 bf16 = 16 MB
    size_t off = (size_t)B_ * HW * C2 * 2;
    float* rnorm = (float*)(ws + off);                   // 128 KB
    float* pmax  = (float*)(ws + off + (128 << 10));     // 128 KB
    int*   pidx  = (int*)  (ws + off + (256 << 10));     // 128 KB
    int*   idx   = (int*)  (ws + off + (384 << 10));     // 32 KB

    norm_kernel<<<(B_ * HW) / 256, 256, 0, stream>>>(x, rnorm);
    prep_kernel<<<B_ * 128, 256, 0, stream>>>(x, rnorm, lnbf);
    copy_kernel<<<(B_ * C_ * HW / 4) / 256, 256, 0, stream>>>(x, out);
    cos_argmax_kernel<<<B_ * 16 * QSPLIT, THREADS, 0, stream>>>(x, flag, lnbf, pmax, pidx);
    argmax_merge_kernel<<<(B_ * PFLAG + 255) / 256, 256, 0, stream>>>(pmax, pidx, idx);
    gather_kernel<<<(B_ * C2 * HW) / 256, 256, 0, stream>>>(x, flag, idx, out);
}
